// HGNet_10754598109729
// MI455X (gfx1250) — matrix operand order repaired and checked by reading, not compiled, not run on hardware
//
#include <hip/hip_runtime.h>
#include <hip/hip_bf16.h>

// ---------------------------------------------------------------------------
// HGNet on MI455X (gfx1250, wave32, WMMA).
//   N=16384 nodes, IN=512, H=256, k=10.
//   Dominant cost: x @ x^T (275 GFLOP) fused with per-row top-10 -> never
//   materializes the 1 GB distance matrix. Split-bf16 (hi+lo) WMMA gives
//   ~fp32 accuracy at bf16 matrix-core rates.
//   __launch_bounds__(256) on the WMMA kernels raises the VGPR budget
//   (8 waves/block, not the default 1024-thread assumption) -> no spills
//   in the per-tile epilogue.
// ---------------------------------------------------------------------------

typedef __bf16 v16bf __attribute__((ext_vector_type(16)));
typedef __bf16 v8bf  __attribute__((ext_vector_type(8)));
typedef float  v8f   __attribute__((ext_vector_type(8)));

#define NN     16384
#define DIN    512
#define DH     256
#define KNN    10
#define LEAKC  0.01f
#define FHUGE  3.402823466e38f

// ---------------- WMMA fragment helpers (ISA 7.12.2 layouts) ---------------
// A-frag (16x32 bf16, row-major source): lane l (l<16 rows 0..15 with K-half 0,
// l>=16 same rows with K-half 8): VGPR0..3 = A[r][kh+0..7], VGPR4..7 = A[r][kh+16..23].
__device__ __forceinline__ v16bf load_a_frag(const __bf16* p) {
  v8bf a0 = *(const v8bf*)(p);
  v8bf a1 = *(const v8bf*)(p + 16);
  v16bf r;
#pragma unroll
  for (int i = 0; i < 8; ++i) { r[i] = a0[i]; r[i + 8] = a1[i]; }
  return r;
}
// B-frag (32x16 bf16): per lane one column, same K order as A-frag:
// elements 0..7 = K kh+0..7, elements 8..15 = K kh+16..23 (kh = 0 or 8).
// Source is "B-transposed" row-major [ncols][K] -> two 16B loads per lane.
__device__ __forceinline__ v16bf load_b_frag(const __bf16* p) {
  v8bf b0 = *(const v8bf*)(p);
  v8bf b1 = *(const v8bf*)(p + 16);
  v16bf r;
#pragma unroll
  for (int i = 0; i < 8; ++i) { r[i] = b0[i]; r[i + 8] = b1[i]; }
  return r;
}

// acc += hi*hi + hi*lo + lo*hi  (drops lo*lo ~ 2^-16 rel)
__device__ __forceinline__ v8f wmma3(v16bf ah, v16bf al, v16bf bh, v16bf bl, v8f acc) {
  acc = __builtin_amdgcn_wmma_f32_16x16x32_bf16(false, ah, false, bh, (short)0, acc, false, false);
  acc = __builtin_amdgcn_wmma_f32_16x16x32_bf16(false, ah, false, bl, (short)0, acc, false, false);
  acc = __builtin_amdgcn_wmma_f32_16x16x32_bf16(false, al, false, bh, (short)0, acc, false, false);
  return acc;
}

// ---------------- top-10 insertion (sorted ascending, registers) -----------
__device__ __forceinline__ void insert10(float d, int j, float (&bd)[10], int (&bi)[10]) {
  if (d >= bd[9]) return;
#pragma unroll
  for (int t = 9; t > 0; --t) {
    if (d < bd[t - 1]) { bd[t] = bd[t - 1]; bi[t] = bi[t - 1]; }
    else               { bd[t] = d;         bi[t] = j;         return; }
  }
  bd[0] = d; bi[0] = j;
}

// ---------------- prep: split x into bf16 hi/lo + row norms ----------------
__global__ void __launch_bounds__(256)
prep_x_kernel(const float* __restrict__ x, __bf16* __restrict__ xhi,
              __bf16* __restrict__ xlo, float* __restrict__ x2) {
  const int row = blockIdx.x, tid = threadIdx.x;
  float s = 0.f;
  for (int c = tid; c < DIN; c += 256) {
    float v = x[(size_t)row * DIN + c];
    __bf16 h = (__bf16)v;
    __bf16 l = (__bf16)(v - (float)h);
    xhi[(size_t)row * DIN + c] = h;
    xlo[(size_t)row * DIN + c] = l;
    s += v * v;
  }
  __shared__ float red[256];
  red[tid] = s; __syncthreads();
  for (int off = 128; off > 0; off >>= 1) {
    if (tid < off) red[tid] += red[tid + off];
    __syncthreads();
  }
  if (tid == 0) x2[row] = red[0];
}

// transpose + split theta:  src[K][Ncols] -> dstT[Ncols][K] (bf16 hi/lo)
__global__ void __launch_bounds__(256)
tsplit_kernel(const float* __restrict__ src, __bf16* __restrict__ dhi,
              __bf16* __restrict__ dlo, int K, int Ncols) {
  int t = blockIdx.x * 256 + threadIdx.x;
  if (t >= K * Ncols) return;
  int k = t / Ncols, n = t - k * Ncols;
  float v = src[t];
  __bf16 h = (__bf16)v;
  __bf16 l = (__bf16)(v - (float)h);
  dhi[(size_t)n * K + k] = h;
  dlo[(size_t)n * K + k] = l;
}

// ---------------- fused Gram + kNN top-10 ----------------------------------
// block = 256 thr = 8 waves; wave w owns rows [ (blk*8+w)*16 , +16 ) and sweeps
// all 1024 column tiles. Ranking key: x2[j] - 2*dot(i,j)  (row term dropped).
__global__ void __launch_bounds__(256)
knn_kernel(const __bf16* __restrict__ xhi, const __bf16* __restrict__ xlo,
           const float* __restrict__ x2, int* __restrict__ nidx) {
  const int wave = threadIdx.x >> 5;
  const int lane = threadIdx.x & 31;
  const int lrow = lane & 15;
  const int rbase = (blockIdx.x * 8 + wave) * 16;

  const int khA = (lane < 16) ? 0 : 8;    // A-frag K-half
  const int khB = (lane < 16) ? 0 : 8;    // B-frag K-half
  const int rowoff = (lane < 16) ? 0 : 8; // C-frag row offset

  const __bf16* arow_hi = xhi + (size_t)(rbase + lrow) * DIN + khA;
  const __bf16* arow_lo = xlo + (size_t)(rbase + lrow) * DIN + khA;

  __shared__ float tile[8][16][16];
  __shared__ float mD[8][16][10];
  __shared__ int   mI[8][16][10];

  float bd[10]; int bi[10];
#pragma unroll
  for (int m = 0; m < 10; ++m) { bd[m] = FHUGE; bi[m] = 0; }

  const int cbase = (lane >> 4) * 8;      // each half-wave scans 8 cols/row

  for (int jt = 0; jt < NN / 16; ++jt) {
    const int jbase = jt * 16;
    const __bf16* brow_hi = xhi + (size_t)(jbase + lrow) * DIN + khB;
    const __bf16* brow_lo = xlo + (size_t)(jbase + lrow) * DIN + khB;

    v8f acc = {};
#pragma unroll 2
    for (int ks = 0; ks < DIN; ks += 32) {
      v16bf ah = load_a_frag(arow_hi + ks);
      v16bf al = load_a_frag(arow_lo + ks);
      v16bf bh = load_b_frag(brow_hi + ks);
      v16bf bl = load_b_frag(brow_lo + ks);
      acc = wmma3(ah, al, bh, bl, acc);
    }

    // register tile -> LDS (row-major), keyed distance
    const float x2col = x2[jbase + lrow];
#pragma unroll
    for (int r = 0; r < 8; ++r)
      tile[wave][rowoff + r][lrow] = x2col - 2.0f * acc[r];
    asm volatile("s_wait_dscnt 0x0" ::: "memory");   // cross-lane LDS hazard

    // scan: lane handles row lrow, cols cbase..cbase+7
#pragma unroll
    for (int c = 0; c < 8; ++c)
      insert10(tile[wave][lrow][cbase + c], jbase + cbase + c, bd, bi);
  }

  // merge the two half-wave top-10 lists per row
  if (lane >= 16) {
#pragma unroll
    for (int m = 0; m < 10; ++m) { mD[wave][lrow][m] = bd[m]; mI[wave][lrow][m] = bi[m]; }
  }
  asm volatile("s_wait_dscnt 0x0" ::: "memory");
  if (lane < 16) {
#pragma unroll
    for (int m = 0; m < 10; ++m) insert10(mD[wave][lrow][m], mI[wave][lrow][m], bd, bi);
#pragma unroll
    for (int m = 0; m < 10; ++m) nidx[(size_t)(rbase + lrow) * KNN + m] = bi[m];
  }
}

// ---------------- generic split-bf16 GEMM: C[N][M] = A[N][K] @ B (BT[M][K]) --
__global__ void __launch_bounds__(256)
gemm_kernel(const __bf16* __restrict__ Ahi, const __bf16* __restrict__ Alo,
            const __bf16* __restrict__ BThi, const __bf16* __restrict__ BTlo,
            float* __restrict__ C, int Nrows, int K, int M) {
  const int wave = threadIdx.x >> 5;
  const int lane = threadIdx.x & 31;
  const int lrow = lane & 15;
  const int tiles_m = M >> 4;
  const int id = blockIdx.x * 8 + wave;
  if (id >= (Nrows >> 4) * tiles_m) return;
  const int rbase = (id / tiles_m) * 16;
  const int cbase = (id % tiles_m) * 16;

  const int khA = (lane < 16) ? 0 : 8;
  const int khB = (lane < 16) ? 0 : 8;
  const int rowoff = (lane < 16) ? 0 : 8;

  const __bf16* arow_hi = Ahi  + (size_t)(rbase + lrow) * K + khA;
  const __bf16* arow_lo = Alo  + (size_t)(rbase + lrow) * K + khA;
  const __bf16* brow_hi = BThi + (size_t)(cbase + lrow) * K + khB;
  const __bf16* brow_lo = BTlo + (size_t)(cbase + lrow) * K + khB;

  v8f acc = {};
#pragma unroll 2
  for (int ks = 0; ks < K; ks += 32) {
    v16bf ah = load_a_frag(arow_hi + ks);
    v16bf al = load_a_frag(arow_lo + ks);
    v16bf bh = load_b_frag(brow_hi + ks);
    v16bf bl = load_b_frag(brow_lo + ks);
    acc = wmma3(ah, al, bh, bl, acc);
  }
  const int col = cbase + lrow;
#pragma unroll
  for (int r = 0; r < 8; ++r)
    C[(size_t)(rbase + rowoff + r) * M + col] = acc[r];
}

// ---------------- hypergraph message passing --------------------------------
__global__ void __launch_bounds__(256)
deg_kernel(const int* __restrict__ nidx, float* __restrict__ deg) {
  int t = blockIdx.x * 256 + threadIdx.x;
  if (t < NN * KNN) atomicAdd(&deg[nidx[t]], 1.0f);
}

// e[i] = mean over k members of xw
__global__ void __launch_bounds__(256)
edge_mean_kernel(const float* __restrict__ xw, const int* __restrict__ nidx,
                 float* __restrict__ e) {
  const int i = blockIdx.x, c = threadIdx.x;
  __shared__ int nb[KNN];
  if (c < KNN) nb[c] = nidx[(size_t)i * KNN + c];
  __syncthreads();
  float s = 0.f;
#pragma unroll
  for (int m = 0; m < KNN; ++m) s += xw[(size_t)nb[m] * DH + c];
  e[(size_t)i * DH + c] = s * (1.0f / KNN);
}

// agg[v] += e[i] for each member v of hyperedge i
__global__ void __launch_bounds__(256)
node_agg_kernel(const float* __restrict__ e, const int* __restrict__ nidx,
                float* __restrict__ agg) {
  const int i = blockIdx.x, c = threadIdx.x;
  __shared__ int nb[KNN];
  if (c < KNN) nb[c] = nidx[(size_t)i * KNN + c];
  __syncthreads();
  const float v = e[(size_t)i * DH + c];
#pragma unroll
  for (int m = 0; m < KNN; ++m) atomicAdd(&agg[(size_t)nb[m] * DH + c], v);
}

__device__ __forceinline__ float finish_val(const float* agg, const float* deg,
                                            const float* bias, int t) {
  const int v = t >> 8, c = t & (DH - 1);
  float o = agg[t] / fmaxf(deg[v], 1.0f) + bias[c];
  return (o > 0.f) ? o : LEAKC * o;
}

// layer-1 epilogue: leaky + split-bf16 for next GEMM
__global__ void __launch_bounds__(256)
fin_hidden_kernel(const float* __restrict__ agg, const float* __restrict__ deg,
                  const float* __restrict__ bias,
                  __bf16* __restrict__ hhi, __bf16* __restrict__ hlo) {
  int t = blockIdx.x * 256 + threadIdx.x;
  float o = finish_val(agg, deg, bias, t);
  __bf16 h = (__bf16)o;
  hhi[t] = h;
  hlo[t] = (__bf16)(o - (float)h);
}

// layer-2 epilogue: leaky -> f32 output
__global__ void __launch_bounds__(256)
fin_out_kernel(const float* __restrict__ agg, const float* __restrict__ deg,
               const float* __restrict__ bias, float* __restrict__ out) {
  int t = blockIdx.x * 256 + threadIdx.x;
  out[t] = finish_val(agg, deg, bias, t);
}

// ---------------------------------------------------------------------------
extern "C" void kernel_launch(void* const* d_in, const int* in_sizes, int n_in,
                              void* d_out, int out_size, void* d_ws, size_t ws_size,
                              hipStream_t stream) {
  const float* x   = (const float*)d_in[0];
  const float* th1 = (const float*)d_in[1];
  const float* b1  = (const float*)d_in[2];
  const float* th2 = (const float*)d_in[3];
  const float* b2  = (const float*)d_in[4];
  float* out = (float*)d_out;

  // -------- workspace carve (256B aligned) --------
  char* p = (char*)d_ws;
  auto alloc = [&](size_t bytes) -> void* {
    void* r = (void*)p;
    p += (bytes + 255) & ~(size_t)255;
    return r;
  };
  __bf16* xhi  = (__bf16*)alloc((size_t)NN * DIN * 2);
  __bf16* xlo  = (__bf16*)alloc((size_t)NN * DIN * 2);
  float*  x2   = (float*) alloc((size_t)NN * 4);
  __bf16* t1hi = (__bf16*)alloc((size_t)DH * DIN * 2);
  __bf16* t1lo = (__bf16*)alloc((size_t)DH * DIN * 2);
  __bf16* t2hi = (__bf16*)alloc((size_t)DH * DH * 2);
  __bf16* t2lo = (__bf16*)alloc((size_t)DH * DH * 2);
  int*    nidx = (int*)   alloc((size_t)NN * KNN * 4);
  float*  deg  = (float*) alloc((size_t)NN * 4);
  float*  xw   = (float*) alloc((size_t)NN * DH * 4);
  float*  e    = (float*) alloc((size_t)NN * DH * 4);
  float*  agg  = (float*) alloc((size_t)NN * DH * 4);
  __bf16* hhi  = (__bf16*)alloc((size_t)NN * DH * 2);
  __bf16* hlo  = (__bf16*)alloc((size_t)NN * DH * 2);
  (void)ws_size; (void)in_sizes; (void)n_in; (void)out_size;

  const int gemm_blocks = (NN / 16) * (DH / 16) / 8;   // 8 waves/block, 1 tile/wave

  // -------- stage 0: precision split + norms --------
  prep_x_kernel<<<NN, 256, 0, stream>>>(x, xhi, xlo, x2);
  tsplit_kernel<<<(DIN * DH + 255) / 256, 256, 0, stream>>>(th1, t1hi, t1lo, DIN, DH);
  tsplit_kernel<<<(DH * DH + 255) / 256, 256, 0, stream>>>(th2, t2hi, t2lo, DH, DH);

  // -------- stage 1: fused Gram matrix + kNN top-10 --------
  knn_kernel<<<NN / 16 / 8, 256, 0, stream>>>(xhi, xlo, x2, nidx);

  hipMemsetAsync(deg, 0, (size_t)NN * 4, stream);
  deg_kernel<<<(NN * KNN + 255) / 256, 256, 0, stream>>>(nidx, deg);

  // -------- layer 1 --------
  gemm_kernel<<<gemm_blocks, 256, 0, stream>>>(xhi, xlo, t1hi, t1lo, xw, NN, DIN, DH);
  edge_mean_kernel<<<NN, DH, 0, stream>>>(xw, nidx, e);
  hipMemsetAsync(agg, 0, (size_t)NN * DH * 4, stream);
  node_agg_kernel<<<NN, DH, 0, stream>>>(e, nidx, agg);
  fin_hidden_kernel<<<NN * DH / 256, 256, 0, stream>>>(agg, deg, b1, hhi, hlo);

  // -------- layer 2 --------
  gemm_kernel<<<gemm_blocks, 256, 0, stream>>>(hhi, hlo, t2hi, t2lo, xw, NN, DH, DH);
  edge_mean_kernel<<<NN, DH, 0, stream>>>(xw, nidx, e);
  hipMemsetAsync(agg, 0, (size_t)NN * DH * 4, stream);
  node_agg_kernel<<<NN, DH, 0, stream>>>(e, nidx, agg);
  fin_out_kernel<<<NN * DH / 256, 256, 0, stream>>>(agg, deg, b2, out);
}
